// GPT_50182397886758
// MI455X (gfx1250) — compile-verified, matrix operand order checked by reading, not run on hardware
//
#include <hip/hip_runtime.h>
#include <hip/hip_bf16.h>
#include <math.h>

// ---------------------------------------------------------------------------
// GPT-2-ish forward (B=4,T=1024,C=768,NH=12,L=2,V=32000) for gfx1250 (CDNA5).
// All GEMMs + attention matmuls run on v_wmma_f32_16x16x32_bf16 (wave32).
// Weights pre-converted fp32 -> bf16 once. GEMM stages its A-slab into LDS
// with global_load_async_to_lds_b128 (ASYNCcnt) and reads fragments via DS.
// ---------------------------------------------------------------------------

#define B_  4
#define T_  1024
#define C_  768
#define NH_ 12
#define L_  2
#define V_  32000
#define HD_ 64
#define M_  (B_ * T_)   // 4096 token rows

#define KSLAB 256       // K-slab staged in LDS per block (32 rows x 256 bf16)

typedef __attribute__((ext_vector_type(16))) __bf16 bf16x16;
typedef __attribute__((ext_vector_type(8)))  float  floatx8;

union BFrag {
    bf16x16        v;
    unsigned short u[16];
};

__device__ __forceinline__ unsigned short f2bf(float f) {
    unsigned int x = __float_as_uint(f);
    x += 0x7fffu + ((x >> 16) & 1u);   // round-to-nearest-even
    return (unsigned short)(x >> 16);
}

// ---------------------------------------------------------------------------
// fp32 -> bf16 bulk conversion (vectorized: b128 in, b64 out)
// ---------------------------------------------------------------------------
__global__ __launch_bounds__(256) void cvt_bf16_k(
    const float4* __restrict__ in, ushort4* __restrict__ out, long n4)
{
    const long i = (long)blockIdx.x * 256 + threadIdx.x;
    if (i < n4) {
        const float4 f = in[i];
        ushort4 o;
        o.x = f2bf(f.x); o.y = f2bf(f.y); o.z = f2bf(f.z); o.w = f2bf(f.w);
        out[i] = o;
    }
}

// ---------------------------------------------------------------------------
// GEMM: out[M,N] = act( A_bf16[M,K] @ W_bf16[N,K]^T + bias[N] )
// Block = 4 waves = 32x256 output. Per 256-wide K-slab:
//   - stage A[32 x 256] bf16 (16 KB) into LDS via global_load_async_to_lds_b128
//   - inner loop: A fragments from LDS (ds_load_b128), B streamed from global,
//     8 WMMAs per K-step of 32 (2 M-subtiles x 4 N-subtiles per wave).
// A-fragment layout (ISA 7.12.2, 16-bit A 16x32):
//   lanes 0-15 : row M=lane,    K = {k0..k0+7, k0+16..k0+23}
//   lanes 16-31: row M=lane-16, K = {k0+8..k0+15, k0+24..k0+31}
// B mirrors A with N in place of M.
// ---------------------------------------------------------------------------
template <bool GELU, bool OUT_BF16>
__global__ __launch_bounds__(128) void gemm_bf16(
    const unsigned short* __restrict__ A,   // [M,K] bf16
    const unsigned short* __restrict__ W,   // [N,K] bf16
    const float*          __restrict__ bias,// [N] or nullptr
    void*                 __restrict__ outp,// [M,N] f32 or bf16
    int M, int N, int K)
{
    extern __shared__ unsigned short ldsA[];   // [32][KSLAB] bf16 slab
    const int tid  = threadIdx.x;
    const int wave = tid >> 5;
    const int lane = tid & 31;
    const int half = lane >> 4;
    const int lr   = lane & 15;
    const int m0   = blockIdx.y * 32;
    const int n0   = (blockIdx.x * 4 + wave) * 64;

    floatx8 c[2][4];
#pragma unroll
    for (int t = 0; t < 2; ++t)
#pragma unroll
        for (int g = 0; g < 4; ++g)
#pragma unroll
            for (int r = 0; r < 8; ++r) c[t][g][r] = 0.0f;

    const unsigned long gA = (unsigned long)(A + (long)m0 * K);

    for (int ks = 0; ks < K; ks += KSLAB) {
        // ---- async stage: A[32 x KSLAB] -> LDS (1024 x 16B, 8 per lane) ----
#pragma unroll
        for (int it = 0; it < 8; ++it) {
            const int i    = tid + it * 128;
            const int row  = i >> 5;          // 0..31
            const int chnk = i & 31;          // 32 x 16B per row
            const unsigned      lds_off = (unsigned)(row * (KSLAB * 2) + chnk * 16);
            const unsigned long ga      = gA + (unsigned long)row * ((long)K * 2)
                                             + (unsigned long)(ks * 2 + chnk * 16);
            asm volatile("global_load_async_to_lds_b128 %0, %1, off"
                         :: "v"(lds_off), "v"(ga) : "memory");
        }
        asm volatile("s_wait_asynccnt 0x0" ::: "memory");
        __syncthreads();

        __builtin_prefetch(&W[(long)(n0 + lr) * K + ks + KSLAB], 0, 1);

        for (int k0 = 0; k0 < KSLAB; k0 += 32) {
            const int kb = k0 + half * 8;

            BFrag a[2];
#pragma unroll
            for (int t = 0; t < 2; ++t)
#pragma unroll
                for (int i = 0; i < 8; ++i) {
                    a[t].u[i]     = ldsA[(t * 16 + lr) * KSLAB + kb + i];
                    a[t].u[8 + i] = ldsA[(t * 16 + lr) * KSLAB + kb + 16 + i];
                }
#pragma unroll
            for (int g = 0; g < 4; ++g) {
                const long wrow = (long)(n0 + g * 16 + lr) * K + ks;
                BFrag b;
#pragma unroll
                for (int i = 0; i < 8; ++i) {
                    b.u[i]     = W[wrow + kb + i];
                    b.u[8 + i] = W[wrow + kb + 16 + i];
                }
                c[0][g] = __builtin_amdgcn_wmma_f32_16x16x32_bf16(
                              false, a[0].v, false, b.v, (short)0, c[0][g], false, false);
                c[1][g] = __builtin_amdgcn_wmma_f32_16x16x32_bf16(
                              false, a[1].v, false, b.v, (short)0, c[1][g], false, false);
            }
        }
        __syncthreads();   // all DS reads done before next slab overwrites LDS
    }

    // C layout: VGPR r, lanes 0-15 -> M=r, lanes 16-31 -> M=8+r, N = lane%16
#pragma unroll
    for (int t = 0; t < 2; ++t)
#pragma unroll
        for (int g = 0; g < 4; ++g) {
            const int   n  = n0 + g * 16 + lr;
            const float bv = bias ? bias[n] : 0.0f;
#pragma unroll
            for (int r = 0; r < 8; ++r) {
                const int m = m0 + t * 16 + r + half * 8;
                float v = c[t][g][r] + bv;
                if (GELU) v = 0.5f * v * (1.0f + erff(v * 0.70710678118654752f));
                if (OUT_BF16)
                    ((unsigned short*)outp)[(long)m * N + n] = f2bf(v);
                else
                    ((float*)outp)[(long)m * N + n] = v;
            }
        }
}

// ---------------------------------------------------------------------------
// Flash-style causal attention. 1 wave per (b, h, 16-query tile).
// S = Q(16x64) @ K^T          : 2 WMMAs per 16-key group
// O += softmax(S)(16x32) @ V  : 4 WMMAs per 32-key tile (P restaged via LDS)
// ---------------------------------------------------------------------------
__global__ __launch_bounds__(32) void attn_flash(
    const unsigned short* __restrict__ qkv,   // [B,T,3C] bf16
    unsigned short*       __restrict__ o_out) // [B,T,C]  bf16
{
    const int lane = threadIdx.x & 31;
    const int half = lane >> 4;
    const int lr   = lane & 15;
    const int qt   = blockIdx.x;
    const int h    = blockIdx.y;
    const int b    = blockIdx.z;
    const int C3   = 3 * C_;

    __shared__ float p_lds[16 * 32];

    // Q fragments (A-matrix), 2 chunks of K=32 covering HD=64
    BFrag aq[2];
    {
        const long base = ((long)(b * T_ + qt * 16 + lr) * C3) + h * HD_;
#pragma unroll
        for (int ch = 0; ch < 2; ++ch) {
            const int kb = ch * 32 + half * 8;
#pragma unroll
            for (int i = 0; i < 8; ++i) {
                aq[ch].u[i]     = qkv[base + kb + i];
                aq[ch].u[8 + i] = qkv[base + kb + 16 + i];
            }
        }
    }

    floatx8 o[4];
    float mrow[8], lrow[8];
#pragma unroll
    for (int g = 0; g < 4; ++g)
#pragma unroll
        for (int r = 0; r < 8; ++r) o[g][r] = 0.0f;
#pragma unroll
    for (int r = 0; r < 8; ++r) { mrow[r] = -1e30f; lrow[r] = 0.0f; }

    const int nkt = qt / 2 + 1;   // ceil((qt+1)*16 / 32) causal tiles

    for (int kt = 0; kt < nkt; ++kt) {
        const int kbse = kt * 32;

        // ---- S = Q @ K^T over two 16-key groups ----
        floatx8 s[2];
#pragma unroll
        for (int g2 = 0; g2 < 2; ++g2) {
#pragma unroll
            for (int r = 0; r < 8; ++r) s[g2][r] = 0.0f;
            const long kr = ((long)(b * T_ + kbse + g2 * 16 + lr) * C3) + C_ + h * HD_;
#pragma unroll
            for (int ch = 0; ch < 2; ++ch) {
                const int kb = ch * 32 + half * 8;
                BFrag bk;
#pragma unroll
                for (int i = 0; i < 8; ++i) {
                    bk.u[i]     = qkv[kr + kb + i];
                    bk.u[8 + i] = qkv[kr + kb + 16 + i];
                }
                s[g2] = __builtin_amdgcn_wmma_f32_16x16x32_bf16(
                            false, aq[ch].v, false, bk.v, (short)0, s[g2], false, false);
            }
        }

        // ---- scale, causal mask, online softmax (half-wave reductions) ----
#pragma unroll
        for (int r = 0; r < 8; ++r) {
            const int qrow = qt * 16 + r + half * 8;
            float s0 = s[0][r] * 0.125f;            // 1/sqrt(64)
            float s1 = s[1][r] * 0.125f;
            if (kbse + lr      > qrow) s0 = -1e30f;
            if (kbse + 16 + lr > qrow) s1 = -1e30f;

            float v = fmaxf(s0, s1);
            v = fmaxf(v, __shfl_xor(v, 1));
            v = fmaxf(v, __shfl_xor(v, 2));
            v = fmaxf(v, __shfl_xor(v, 4));
            v = fmaxf(v, __shfl_xor(v, 8));         // max within 16-lane half

            const float mn    = fmaxf(mrow[r], v);
            const float alpha = __expf(mrow[r] - mn);
            const float p0    = __expf(s0 - mn);
            const float p1    = __expf(s1 - mn);

            float rs = p0 + p1;
            rs += __shfl_xor(rs, 1);
            rs += __shfl_xor(rs, 2);
            rs += __shfl_xor(rs, 4);
            rs += __shfl_xor(rs, 8);

            lrow[r] = lrow[r] * alpha + rs;
            mrow[r] = mn;
#pragma unroll
            for (int g = 0; g < 4; ++g) o[g][r] *= alpha;

            p_lds[(r + half * 8) * 32 + lr]      = p0;
            p_lds[(r + half * 8) * 32 + 16 + lr] = p1;
        }
        __syncthreads();

        // ---- P restaged to A-fragment layout ----
        BFrag pa;
#pragma unroll
        for (int i = 0; i < 8; ++i) {
            pa.u[i]     = f2bf(p_lds[lr * 32 + half * 8 + i]);
            pa.u[8 + i] = f2bf(p_lds[lr * 32 + 16 + half * 8 + i]);
        }

        // ---- O += P @ V ----
#pragma unroll
        for (int g = 0; g < 4; ++g) {
            BFrag bv;
#pragma unroll
            for (int i = 0; i < 8; ++i) {
                const int j0 = half * 8 + i;
                bv.u[i]     = qkv[((long)(b * T_ + kbse + j0) * C3)
                                  + 2 * C_ + h * HD_ + g * 16 + lr];
                bv.u[8 + i] = qkv[((long)(b * T_ + kbse + 16 + j0) * C3)
                                  + 2 * C_ + h * HD_ + g * 16 + lr];
            }
            o[g] = __builtin_amdgcn_wmma_f32_16x16x32_bf16(
                       false, pa.v, false, bv.v, (short)0, o[g], false, false);
        }
        __syncthreads();
    }

    // ---- epilogue: O / l, store bf16 ----
#pragma unroll
    for (int g = 0; g < 4; ++g)
#pragma unroll
        for (int r = 0; r < 8; ++r) {
            const float v = o[g][r] / lrow[r];
            o_out[((long)(b * T_ + qt * 16 + r + half * 8) * C_)
                  + h * HD_ + g * 16 + lr] = f2bf(v);
        }
}

// ---------------------------------------------------------------------------
// LayerNorm over C=768; optionally two (g,b) sets in one pass. bf16 outputs.
// ---------------------------------------------------------------------------
__global__ __launch_bounds__(256) void ln_dual(
    const float* __restrict__ x,
    const float* __restrict__ g1, const float* __restrict__ b1,
    unsigned short* __restrict__ o1,
    const float* __restrict__ g2, const float* __restrict__ b2,
    unsigned short* __restrict__ o2)
{
    const long  row = blockIdx.x;
    const float* xr = x + row * C_;
    __shared__ float sh[8], sh2[8], stat[2];

    float s = 0.0f, s2 = 0.0f;
    for (int c = threadIdx.x; c < C_; c += 256) {
        const float v = xr[c];
        s += v; s2 += v * v;
    }
    s  += __shfl_xor(s, 1);  s2 += __shfl_xor(s2, 1);
    s  += __shfl_xor(s, 2);  s2 += __shfl_xor(s2, 2);
    s  += __shfl_xor(s, 4);  s2 += __shfl_xor(s2, 4);
    s  += __shfl_xor(s, 8);  s2 += __shfl_xor(s2, 8);
    s  += __shfl_xor(s, 16); s2 += __shfl_xor(s2, 16);
    const int w = threadIdx.x >> 5;
    if ((threadIdx.x & 31) == 0) { sh[w] = s; sh2[w] = s2; }
    __syncthreads();
    if (threadIdx.x == 0) {
        float ts = 0.0f, ts2 = 0.0f;
        for (int i = 0; i < 8; ++i) { ts += sh[i]; ts2 += sh2[i]; }
        const float mu  = ts / C_;
        const float var = ts2 / C_ - mu * mu;
        stat[0] = mu;
        stat[1] = rsqrtf(var + 1e-5f);
    }
    __syncthreads();
    const float mu = stat[0], rstd = stat[1];
    for (int c = threadIdx.x; c < C_; c += 256) {
        const float n = (xr[c] - mu) * rstd;
        o1[row * C_ + c] = f2bf(n * g1[c] + b1[c]);
        if (o2) o2[row * C_ + c] = f2bf(n * g2[c] + b2[c]);
    }
}

// ---------------------------------------------------------------------------
__global__ __launch_bounds__(256) void embed_k(
    const int* __restrict__ idx, const float* __restrict__ wte,
    const float* __restrict__ wpe, float* __restrict__ x)
{
    const int  bt   = blockIdx.x;
    const int  t    = bt % T_;
    const long wrow = (long)idx[bt] * C_;
    float* xr = x + (long)bt * C_;
    const float* pr = wpe + (long)t * C_;
    for (int c = threadIdx.x; c < C_; c += 256)
        xr[c] = wte[wrow + c] + pr[c];
}

__global__ __launch_bounds__(256) void add2_k(
    float* __restrict__ x, const float* __restrict__ a,
    const float* __restrict__ b, long n)
{
    const long i = (long)blockIdx.x * blockDim.x + threadIdx.x;
    if (i < n) x[i] += a[i] + b[i];
}

// ---------------------------------------------------------------------------
extern "C" void kernel_launch(void* const* d_in, const int* in_sizes, int n_in,
                              void* d_out, int out_size, void* d_ws, size_t ws_size,
                              hipStream_t stream)
{
    (void)in_sizes; (void)n_in; (void)out_size; (void)ws_size;

    const int*   idx    = (const int*)  d_in[0];
    const float* wte    = (const float*)d_in[1];
    const float* wpe    = (const float*)d_in[2];
    const float* ln1_g  = (const float*)d_in[3];
    const float* ln1_b  = (const float*)d_in[4];
    const float* ln2_g  = (const float*)d_in[5];
    const float* ln2_b  = (const float*)d_in[6];
    const float* qkv_w  = (const float*)d_in[7];
    const float* qkv_b  = (const float*)d_in[8];
    const float* proj_w = (const float*)d_in[9];
    const float* proj_b = (const float*)d_in[10];
    const float* fc1_w  = (const float*)d_in[11];
    const float* fc1_b  = (const float*)d_in[12];
    const float* fc2_w  = (const float*)d_in[13];
    const float* fc2_b  = (const float*)d_in[14];
    const float* lnf_g  = (const float*)d_in[15];
    const float* lnf_b  = (const float*)d_in[16];
    float* logits = (float*)d_out;

    // bump-allocate scratch
    char*  ws  = (char*)d_ws;
    size_t off = 0;
    auto alloc = [&](size_t bytes) -> void* {
        void* p = ws + off;
        off += (bytes + 255) & ~(size_t)255;
        return p;
    };
    // activations
    float*          x    = (float*)         alloc((size_t)M_ * C_ * 4);
    unsigned short* h1   = (unsigned short*)alloc((size_t)M_ * C_ * 2);
    unsigned short* h2   = (unsigned short*)alloc((size_t)M_ * C_ * 2);
    unsigned short* qkv  = (unsigned short*)alloc((size_t)M_ * 3 * C_ * 2);
    unsigned short* atto = (unsigned short*)alloc((size_t)M_ * C_ * 2);
    float*          ad   = (float*)         alloc((size_t)M_ * C_ * 4);
    float*          md   = (float*)         alloc((size_t)M_ * C_ * 4);
    unsigned short* mbf  = (unsigned short*)alloc((size_t)M_ * 4 * C_ * 2);
    unsigned short* lnf  = (unsigned short*)alloc((size_t)M_ * C_ * 2);
    // bf16 weights
    const long n_wte  = (long)V_ * C_;
    const long n_qkvw = (long)L_ * 3 * C_ * C_;
    const long n_prjw = (long)L_ * C_ * C_;
    const long n_fc1w = (long)L_ * 4 * C_ * C_;
    const long n_fc2w = (long)L_ * C_ * 4 * C_;
    unsigned short* wteb  = (unsigned short*)alloc((size_t)n_wte  * 2);
    unsigned short* qkvwb = (unsigned short*)alloc((size_t)n_qkvw * 2);
    unsigned short* prjwb = (unsigned short*)alloc((size_t)n_prjw * 2);
    unsigned short* fc1wb = (unsigned short*)alloc((size_t)n_fc1w * 2);
    unsigned short* fc2wb = (unsigned short*)alloc((size_t)n_fc2w * 2);

    // one-time (per launch) weight conversion fp32 -> bf16
    auto cvt = [&](const float* src, unsigned short* dst, long n) {
        const long n4 = n / 4;
        cvt_bf16_k<<<(unsigned)((n4 + 255) / 256), 256, 0, stream>>>(
            (const float4*)src, (ushort4*)dst, n4);
    };
    cvt(wte,    wteb,  n_wte);
    cvt(qkv_w,  qkvwb, n_qkvw);
    cvt(proj_w, prjwb, n_prjw);
    cvt(fc1_w,  fc1wb, n_fc1w);
    cvt(fc2_w,  fc2wb, n_fc2w);

    embed_k<<<M_, 256, 0, stream>>>(idx, wte, wpe, x);

    const unsigned slab_bytes = 32 * KSLAB * 2;   // 16 KB LDS per block

    for (int l = 0; l < L_; ++l) {
        ln_dual<<<M_, 256, 0, stream>>>(x,
            ln1_g + (long)l * C_, ln1_b + (long)l * C_, h1,
            ln2_g + (long)l * C_, ln2_b + (long)l * C_, h2);

        // qkv: [4096,768] x [2304,768]^T -> bf16
        gemm_bf16<false, true><<<dim3(3 * C_ / 256, M_ / 32), 128, slab_bytes, stream>>>(
            h1, qkvwb + (long)l * 3 * C_ * C_, qkv_b + (long)l * 3 * C_,
            qkv, M_, 3 * C_, C_);

        attn_flash<<<dim3(T_ / 16, NH_, B_), 32, 0, stream>>>(qkv, atto);

        // proj: [4096,768] x [768,768]^T -> f32 delta
        gemm_bf16<false, false><<<dim3(C_ / 256, M_ / 32), 128, slab_bytes, stream>>>(
            atto, prjwb + (long)l * C_ * C_, proj_b + (long)l * C_,
            ad, M_, C_, C_);

        // fc1 + GELU: [4096,768] x [3072,768]^T -> bf16
        gemm_bf16<true, true><<<dim3(4 * C_ / 256, M_ / 32), 128, slab_bytes, stream>>>(
            h2, fc1wb + (long)l * 4 * C_ * C_, fc1_b + (long)l * 4 * C_,
            mbf, M_, 4 * C_, C_);

        // fc2: [4096,3072] x [768,3072]^T -> f32 delta
        gemm_bf16<false, false><<<dim3(C_ / 256, M_ / 32), 128, slab_bytes, stream>>>(
            mbf, fc2wb + (long)l * C_ * 4 * C_, fc2_b + (long)l * C_,
            md, M_, C_, 4 * C_);

        add2_k<<<(unsigned)(((long)M_ * C_ + 255) / 256), 256, 0, stream>>>(
            x, ad, md, (long)M_ * C_);
    }

    ln_dual<<<M_, 256, 0, stream>>>(x, lnf_g, lnf_b, lnf,
                                    nullptr, nullptr, nullptr);

    // lm head: [4096,768] x [32000,768]^T -> f32 logits (tied bf16 wte, no bias)
    gemm_bf16<false, false><<<dim3(V_ / 256, M_ / 32), 128, slab_bytes, stream>>>(
        lnf, wteb, nullptr, logits, M_, V_, C_);
}
